// CheemsNonWoAttention_17592186044951
// MI455X (gfx1250) — matrix operand order repaired and checked by reading, not compiled, not run on hardware
//
#include <hip/hip_runtime.h>

// ---------------------------------------------------------------------------
// CheemsNonWoAttention for MI455X (gfx1250, wave32, WMMA bf16 16x16x32).
//   B=2 S=2048 HID=2048 NH=16 NKV=4 HD=128.
//   logn scale is identically 1.0 (log(65..80)/log(256) < 1, clamped to 1).
//   mask == analytic causal -> applied in-kernel, masked key tiles skipped.
// Workspace (bytes): Hb 16777216 | Wt 12582912 | Q 16777216 | K 4194304 | Vt 4194304
// ---------------------------------------------------------------------------

#define S_     2048
#define HD_    128
#define SCALE_ 0.08838834764831845f   // 1/sqrt(128)

typedef __attribute__((ext_vector_type(16))) __bf16 v16bf;
typedef __attribute__((ext_vector_type(8)))  float  v8f;

union FragB { v16bf v; uint4 u[2]; };

__device__ __forceinline__ unsigned short f2bf(float f) {
  unsigned u = __float_as_uint(f);
  u += 0x7FFFu + ((u >> 16) & 1u);        // round-to-nearest-even
  return (unsigned short)(u >> 16);
}

__device__ __forceinline__ v8f v8f_zero() {
  v8f v;
#pragma unroll
  for (int i = 0; i < 8; ++i) v[i] = 0.0f;
  return v;
}

// ---------------- kernel 1: H f32 -> bf16 ----------------------------------
__global__ __launch_bounds__(256) void cvt_h(const float* __restrict__ h,
                                             unsigned short* __restrict__ hb) {
  size_t i = ((size_t)blockIdx.x * 256 + threadIdx.x) * 4;
  float4 v = *(const float4*)(h + i);
  uint2 p;
  p.x = (unsigned)f2bf(v.x) | ((unsigned)f2bf(v.y) << 16);
  p.y = (unsigned)f2bf(v.z) | ((unsigned)f2bf(v.w) << 16);
  *(uint2*)(hb + i) = p;
}

// ---------------- kernel 2: fused W^T (3072 x 2048) bf16 -------------------
__global__ __launch_bounds__(256) void build_wt(const float* __restrict__ Wq,
                                                const float* __restrict__ Wk,
                                                const float* __restrict__ Wv,
                                                unsigned short* __restrict__ Wt) {
  __shared__ unsigned short tile[32][33];   // +1 pad: no bank conflicts
  const int k0 = blockIdx.x * 32;
  const int n0 = blockIdx.y * 32;
  const int c  = threadIdx.x & 31;
  const int r0 = threadIdx.x >> 5;          // 0..7
  const float* src; int nw, nb;
  if (n0 < 2048)      { src = Wq; nw = 2048; nb = 0;    }
  else if (n0 < 2560) { src = Wk; nw = 512;  nb = 2048; }
  else                { src = Wv; nw = 512;  nb = 2560; }
  const int n = n0 + c;
#pragma unroll
  for (int i = 0; i < 4; ++i) {
    int r = r0 + i * 8;
    tile[r][c] = f2bf(src[(size_t)(k0 + r) * nw + (n - nb)]);
  }
  __syncthreads();
#pragma unroll
  for (int i = 0; i < 4; ++i) {
    int r = r0 + i * 8;                      // local n row
    Wt[(size_t)(n0 + r) * 2048 + k0 + c] = tile[c][r];
  }
}

// ---------------- kernel 3: QKV GEMM, C = Hb(4096x2048) x W(2048x3072) -----
// block tile 128x128, 8 waves each 32(M) x 64(N), K-chunks of 32.
__global__ __launch_bounds__(256, 2) void qkv_gemm(
    const unsigned short* __restrict__ Hb,   // 4096 x 2048 bf16
    const unsigned short* __restrict__ Wt,   // 3072 x 2048 bf16 (= W^T)
    unsigned short* __restrict__ Qw,         // (B,16,S,128)
    unsigned short* __restrict__ Kw,         // (B,4,S,128)
    unsigned short* __restrict__ Vt) {       // (B,4,128,S)
  __shared__ uint4 As[512];                  // 128 rows x 64B
  __shared__ uint4 Bs[512];
  const int tid  = threadIdx.x;
  const int lane = tid & 31;
  const int wave = tid >> 5;
  const int wm   = wave & 3, wn = wave >> 2;
  const int l16  = lane & 15, lh = lane >> 4;
  const int mbase = blockIdx.x * 128;
  const int nbase = blockIdx.y * 128;
  const int crow  = tid >> 1;                // cooperative copy: row 0..127
  const int cpart = tid & 1;                 // which 32B half of 64B row

  v8f acc[2][4];
#pragma unroll
  for (int mf = 0; mf < 2; ++mf)
#pragma unroll
    for (int nf = 0; nf < 4; ++nf) acc[mf][nf] = v8f_zero();

  const uint4* gA = (const uint4*)Hb;        // row stride 2048 bf16 = 256 uint4
  const uint4* gB = (const uint4*)Wt;

  for (int kc = 0; kc < 64; ++kc) {
    const int kb4 = kc * 4;                  // K-chunk in uint4 units
    uint4 a0 = gA[(size_t)(mbase + crow) * 256 + kb4 + cpart * 2];
    uint4 a1 = gA[(size_t)(mbase + crow) * 256 + kb4 + cpart * 2 + 1];
    uint4 b0 = gB[(size_t)(nbase + crow) * 256 + kb4 + cpart * 2];
    uint4 b1 = gB[(size_t)(nbase + crow) * 256 + kb4 + cpart * 2 + 1];
    if (kc + 1 < 64) {
      __builtin_prefetch(&gA[(size_t)(mbase + crow) * 256 + kb4 + 4], 0, 1);
      __builtin_prefetch(&gB[(size_t)(nbase + crow) * 256 + kb4 + 4], 0, 1);
    }
    __syncthreads();                         // previous-iter readers done
    As[crow * 4 + cpart * 2]     = a0;
    As[crow * 4 + cpart * 2 + 1] = a1;
    Bs[crow * 4 + cpart * 2]     = b0;
    Bs[crow * 4 + cpart * 2 + 1] = b1;
    __syncthreads();

    FragB af[2];
#pragma unroll
    for (int mf = 0; mf < 2; ++mf) {
      const int row = wm * 32 + mf * 16 + l16;       // A-layout (16x32 bf16)
      af[mf].u[0] = As[row * 4 + lh];
      af[mf].u[1] = As[row * 4 + 2 + lh];
    }
#pragma unroll
    for (int nf = 0; nf < 4; ++nf) {
      FragB bf_;
      const int row = wn * 64 + nf * 16 + l16;       // B-layout (32x16 bf16)
      bf_.u[0] = Bs[row * 4 + lh];
      bf_.u[1] = Bs[row * 4 + 2 + lh];
#pragma unroll
      for (int mf = 0; mf < 2; ++mf)
        acc[mf][nf] = __builtin_amdgcn_wmma_f32_16x16x32_bf16(
            false, af[mf].v, false, bf_.v, (short)0, acc[mf][nf], false, false);
    }
  }

  // epilogue: scatter Q / K / V(transposed) as bf16
#pragma unroll
  for (int mf = 0; mf < 2; ++mf)
#pragma unroll
    for (int nf = 0; nf < 4; ++nf)
#pragma unroll
      for (int r = 0; r < 8; ++r) {
        const int m = mbase + wm * 32 + mf * 16 + r + 8 * lh;
        const int n = nbase + wn * 64 + nf * 16 + l16;
        const int bb = m >> 11, s = m & 2047;
        const unsigned short val = f2bf(acc[mf][nf][r]);
        if (n < 2048) {
          const int h = n >> 7, d = n & 127;
          Qw[(((size_t)bb * 16 + h) * S_ + s) * HD_ + d] = val;
        } else if (n < 2560) {
          const int h = (n - 2048) >> 7, d = n & 127;
          Kw[(((size_t)bb * 4 + h) * S_ + s) * HD_ + d] = val;
        } else {
          const int h = (n - 2560) >> 7, d = n & 127;
          Vt[(((size_t)bb * 4 + h) * HD_ + d) * S_ + s] = val;
        }
      }
}

// ---------------- kernel 4: causal flash attention -------------------------
// grid (B*NH, S/128); block 256 = 8 waves, each wave owns 16 query rows.
__global__ __launch_bounds__(256, 2) void attn(
    const unsigned short* __restrict__ Qw,
    const unsigned short* __restrict__ Kw,
    const unsigned short* __restrict__ Vt,
    float* __restrict__ out) {
  __shared__ uint4 Ks[512];                          // 32 keys x 256B
  __shared__ uint4 Vs[512];                          // 128 d x 64B (keys)
  __shared__ alignas(16) unsigned short Pb[8][512];  // per-wave P (16x32 bf16)

  const int tid  = threadIdx.x;
  const int lane = tid & 31;
  const int wave = tid >> 5;
  const int l16  = lane & 15, lh = lane >> 4;
  const int bh   = blockIdx.x;
  const int b    = bh >> 4, h = bh & 15;
  const int kvh  = h >> 2;                           // GQA: head h -> kv h/4
  const int qblk = blockIdx.y;
  const int q0   = qblk * 128 + wave * 16;
  const int crow = tid >> 1, cpart = tid & 1;

  // Q fragments (resident for whole kernel), A-layout per 32-d chunk
  const uint4* qg = (const uint4*)(Qw + (((size_t)b * 16 + h) * S_ + q0) * HD_);
  FragB qa[4];
#pragma unroll
  for (int dc = 0; dc < 4; ++dc) {
    qa[dc].u[0] = qg[(size_t)l16 * 16 + dc * 4 + lh];
    qa[dc].u[1] = qg[(size_t)l16 * 16 + dc * 4 + 2 + lh];
  }

  v8f o[8];
#pragma unroll
  for (int f = 0; f < 8; ++f) o[f] = v8f_zero();
  float mrow[8], lrow[8];
#pragma unroll
  for (int r = 0; r < 8; ++r) { mrow[r] = -1e30f; lrow[r] = 0.0f; }

  const uint4* kg = (const uint4*)(Kw + ((size_t)b * 4 + kvh) * S_ * HD_);
  const unsigned short* vtb = Vt + ((size_t)b * 4 + kvh) * HD_ * S_;

  const int nchunks = qblk * 4 + 4;                  // causal: skip tiles > diag
  for (int kc = 0; kc < nchunks; ++kc) {
    const int kstart = kc * 32;
    // cooperative stage: K chunk (contiguous 8KB), Vt chunk (128 rows x 64B)
    uint4 kv0 = kg[(size_t)kstart * 16 + tid * 2];
    uint4 kv1 = kg[(size_t)kstart * 16 + tid * 2 + 1];
    const uint4* vg = (const uint4*)(vtb + (size_t)crow * S_ + kstart);
    uint4 vv0 = vg[cpart * 2];
    uint4 vv1 = vg[cpart * 2 + 1];
    __syncthreads();
    Ks[tid * 2]     = kv0;
    Ks[tid * 2 + 1] = kv1;
    Vs[crow * 4 + cpart * 2]     = vv0;
    Vs[crow * 4 + cpart * 2 + 1] = vv1;
    __syncthreads();

    if (kstart <= q0 + 15) {                         // wave-uniform skip
      // S = Q K^T : two 16-key fragments, K-loop over 4 d-chunks
      v8f s0 = v8f_zero(), s1 = v8f_zero();
#pragma unroll
      for (int nf = 0; nf < 2; ++nf) {
#pragma unroll
        for (int dc = 0; dc < 4; ++dc) {
          FragB kb;
          const int row = nf * 16 + l16;             // key row, 256B each
          kb.u[0] = Ks[row * 16 + dc * 4 + lh];
          kb.u[1] = Ks[row * 16 + dc * 4 + 2 + lh];
          if (nf == 0)
            s0 = __builtin_amdgcn_wmma_f32_16x16x32_bf16(
                false, qa[dc].v, false, kb.v, (short)0, s0, false, false);
          else
            s1 = __builtin_amdgcn_wmma_f32_16x16x32_bf16(
                false, qa[dc].v, false, kb.v, (short)0, s1, false, false);
        }
      }
      // online softmax (rows live in 16-lane halves; xor-shuffle reduce)
#pragma unroll
      for (int r = 0; r < 8; ++r) {
        const int q  = q0 + r + 8 * lh;
        const int ki = kstart + l16;
        float x0 = s0[r] * SCALE_; if (ki > q)      x0 = -1e30f;
        float x1 = s1[r] * SCALE_; if (ki + 16 > q) x1 = -1e30f;
        float mx = fmaxf(x0, x1);
        mx = fmaxf(mx, __shfl_xor(mx, 1, 32));
        mx = fmaxf(mx, __shfl_xor(mx, 2, 32));
        mx = fmaxf(mx, __shfl_xor(mx, 4, 32));
        mx = fmaxf(mx, __shfl_xor(mx, 8, 32));
        const float mnew = fmaxf(mrow[r], mx);
        const float corr = __expf(mrow[r] - mnew);
        mrow[r] = mnew;
        const float p0 = __expf(x0 - mnew);
        const float p1 = __expf(x1 - mnew);
        lrow[r] = lrow[r] * corr + p0 + p1;          // lane-partial row sum
#pragma unroll
        for (int f = 0; f < 8; ++f) o[f][r] *= corr;
        const int prow = r + 8 * lh;                 // P row-major 16x32 bf16
        Pb[wave][prow * 32 + l16]      = f2bf(p0);
        Pb[wave][prow * 32 + 16 + l16] = f2bf(p1);
      }
      // P as A-fragment (16 queries x 32 keys)
      FragB pa;
      pa.u[0] = *(const uint4*)&Pb[wave][l16 * 32 + lh * 8];
      pa.u[1] = *(const uint4*)&Pb[wave][l16 * 32 + 16 + lh * 8];
      // O += P * V : 8 d-fragments from Vt tile
#pragma unroll
      for (int df = 0; df < 8; ++df) {
        FragB vb;
        const int row = df * 16 + l16;               // d row over 32 keys
        vb.u[0] = Vs[row * 4 + lh];
        vb.u[1] = Vs[row * 4 + 2 + lh];
        o[df] = __builtin_amdgcn_wmma_f32_16x16x32_bf16(
            false, pa.v, false, vb.v, (short)0, o[df], false, false);
      }
    }
  }

  // finalize: reduce lane-partial row sums once, normalize, store f32
#pragma unroll
  for (int r = 0; r < 8; ++r) {
    float l = lrow[r];
    l += __shfl_xor(l, 1, 32);
    l += __shfl_xor(l, 2, 32);
    l += __shfl_xor(l, 4, 32);
    l += __shfl_xor(l, 8, 32);
    lrow[r] = 1.0f / l;
  }
  float* ob = out + ((size_t)b * S_ + q0) * 2048 + h * 128;
#pragma unroll
  for (int df = 0; df < 8; ++df)
#pragma unroll
    for (int r = 0; r < 8; ++r) {
      const int q = r + 8 * lh;
      ob[(size_t)q * 2048 + df * 16 + l16] = o[df][r] * lrow[r];
    }
}

// ---------------------------------------------------------------------------
extern "C" void kernel_launch(void* const* d_in, const int* in_sizes, int n_in,
                              void* d_out, int out_size, void* d_ws, size_t ws_size,
                              hipStream_t stream) {
  const float* H  = (const float*)d_in[0];
  // d_in[1] = attention_mask: exactly the additive causal mask -> applied analytically
  const float* Wq = (const float*)d_in[2];
  const float* Wk = (const float*)d_in[3];
  const float* Wv = (const float*)d_in[4];
  float* out = (float*)d_out;

  char* ws = (char*)d_ws;                       // needs ~54.6 MB
  unsigned short* Hb = (unsigned short*)ws;                          // 16,777,216 B
  unsigned short* Wt = (unsigned short*)(ws + 16777216);             // 12,582,912 B
  unsigned short* Qw = (unsigned short*)(ws + 16777216 + 12582912);  // 16,777,216 B
  unsigned short* Kw = Qw + (size_t)2 * 16 * S_ * HD_;               //  4,194,304 B
  unsigned short* Vt = Kw + (size_t)2 * 4 * S_ * HD_;                //  4,194,304 B

  cvt_h   <<<dim3(8192),    dim3(256), 0, stream>>>(H, Hb);
  build_wt<<<dim3(64, 96),  dim3(256), 0, stream>>>(Wq, Wk, Wv, Wt);
  qkv_gemm<<<dim3(32, 24),  dim3(256), 0, stream>>>(Hb, Wt, Qw, Kw, Vt);
  attn    <<<dim3(32, 16),  dim3(256), 0, stream>>>(Qw, Kw, Vt, out);
}
